// SyncQwen3VLMoeSparseMoeBlock_44418551775998
// MI455X (gfx1250) — matrix operand order repaired and checked by reading, not compiled, not run on hardware
//
#include <hip/hip_runtime.h>

// ---- Problem constants (from reference) ----
constexpr int B_ = 2, S_ = 1024, H_ = 2048, E_ = 32, I_ = 768, TOPK = 4;
constexpr int T_ = B_ * S_;       // 2048 tokens
constexpr int TK = T_ * TOPK;     // 8192 (token, expert) pairs

// ---- GEMM tiling ----
constexpr int BM = 64, BN = 64, BK = 32;

typedef __attribute__((ext_vector_type(16))) __bf16 v16bf;
typedef __attribute__((ext_vector_type(8)))  float  v8f;
typedef __attribute__((ext_vector_type(4)))  int    i4;

union FragAB { v16bf v; uint4 q[2]; };

#define AS1 __attribute__((address_space(1)))
#define AS3 __attribute__((address_space(3)))

#if __has_builtin(__builtin_amdgcn_global_load_async_to_lds_b128)
#define HAS_ASYNC_LDS 1
#endif

// Use the hardware packed convert (ISA references V_CVT_PK_* family on CDNA5).
#define USE_CVT_PK_ASM 1

__device__ __forceinline__ unsigned short f2bf(float f) {
  unsigned int u = __float_as_uint(f);
  u += 0x7FFFu + ((u >> 16) & 1u);   // RNE
  return (unsigned short)(u >> 16);
}
// pack two fp32 -> two bf16 in one dword
__device__ __forceinline__ unsigned pk2(float a, float b) {
#if __has_builtin(__builtin_amdgcn_cvt_pk_bf16_f32)
  auto r = __builtin_amdgcn_cvt_pk_bf16_f32(a, b);
  return __builtin_bit_cast(unsigned, r);
#elif USE_CVT_PK_ASM
  unsigned r;
  asm("v_cvt_pk_bf16_f32 %0, %1, %2" : "=v"(r) : "v"(a), "v"(b));
  return r;
#else
  return (unsigned)f2bf(a) | ((unsigned)f2bf(b) << 16);
#endif
}
__device__ __forceinline__ void wait_async0() {
#ifdef HAS_ASYNC_LDS
#if __has_builtin(__builtin_amdgcn_s_wait_asynccnt)
  __builtin_amdgcn_s_wait_asynccnt(0);
#else
  asm volatile("s_wait_asynccnt 0x0" ::: "memory");
#endif
#endif
}

// ---- workspace layout (bytes) ----
constexpr size_t OFF_TOPI    = 0;
constexpr size_t OFF_TOPV    = OFF_TOPI    + (size_t)TK * 4;
constexpr size_t OFF_COUNTS  = OFF_TOPV    + (size_t)TK * 4;
constexpr size_t OFF_OFFS    = OFF_COUNTS  + 256;
constexpr size_t OFF_PAIRPOS = OFF_OFFS    + 256;
constexpr size_t OFF_TOKIDX  = OFF_PAIRPOS + (size_t)TK * 4;
constexpr size_t OFF_TOKW    = OFF_TOKIDX  + (size_t)TK * 4;
constexpr size_t OFF_ACT     = OFF_TOKW    + (size_t)TK * 4;               // TK*I bf16
constexpr size_t OFF_PAIROUT = OFF_ACT     + (size_t)TK * I_ * 2;          // TK*H f32
constexpr size_t WS_NEED     = OFF_PAIROUT + (size_t)TK * H_ * 4;

__device__ __forceinline__ float wave_max(float v) {
#pragma unroll
  for (int o = 16; o > 0; o >>= 1) v = fmaxf(v, __shfl_xor(v, o, 32));
  return v;
}
__device__ __forceinline__ float wave_sum(float v) {
#pragma unroll
  for (int o = 16; o > 0; o >>= 1) v += __shfl_xor(v, o, 32);
  return v;
}

// ---------------- Router: one wave32 per token, lane == expert ----------------
__global__ void router_kernel(const float* __restrict__ x,
                              const float* __restrict__ gate_w,
                              int* __restrict__ topi, float* __restrict__ topv,
                              int* __restrict__ counts) {
  const int lane = threadIdx.x & 31;
  const int wv   = threadIdx.x >> 5;
  const int t    = blockIdx.x * 8 + wv;
  if (t >= T_) return;

  const float4* xr = (const float4*)(x + (size_t)t * H_);
  const float4* gr = (const float4*)(gate_w + (size_t)lane * H_);
  float acc = 0.f;
  for (int h = 0; h < H_ / 4; ++h) {
    float4 a = xr[h], b = gr[h];
    acc += a.x * b.x + a.y * b.y + a.z * b.z + a.w * b.w;
  }
  float m = wave_max(acc);
  float p = __expf(acc - m);
  float s = wave_sum(p);
  float prob = p / s;

  float cur = prob;
  int ti[TOPK]; float tv[TOPK];
#pragma unroll
  for (int k = 0; k < TOPK; ++k) {
    float mv = wave_max(cur);
    unsigned long long bal = __ballot(cur == mv);
    int sel = __ffsll(bal) - 1;
    ti[k] = sel; tv[k] = mv;
    if (lane == sel) cur = -1.f;
  }
  float s4 = tv[0] + tv[1] + tv[2] + tv[3];
  if (lane == 0) {
#pragma unroll
    for (int k = 0; k < TOPK; ++k) {
      topi[t * TOPK + k] = ti[k];
      topv[t * TOPK + k] = tv[k] / s4;
      atomicAdd(&counts[ti[k]], 1);
    }
  }
}

// ---------------- Exclusive scan over 32 expert counts ----------------
__global__ void scan_kernel(const int* __restrict__ counts, int* __restrict__ offsets) {
  if (threadIdx.x == 0) {
    int run = 0;
    for (int e = 0; e < E_; ++e) { offsets[e] = run; run += counts[e]; }
  }
}

// ------- Deterministic grouping: one block per expert, counted scatter -------
__global__ void scatter_kernel(const int* __restrict__ topi, const float* __restrict__ topv,
                               const int* __restrict__ offsets,
                               int* __restrict__ tokIdx, float* __restrict__ tokW,
                               int* __restrict__ pairPos) {
  const int e   = blockIdx.x;
  const int tid = threadIdx.x;
  constexpr int PER = TK / 256;            // 32 pairs per thread
  __shared__ int cnt[256];
  const int base = tid * PER;
  int c = 0;
  for (int j = 0; j < PER; ++j) c += (topi[base + j] == e);
  cnt[tid] = c;
  __syncthreads();
  int pre = 0;
  for (int j = 0; j < tid; ++j) pre += cnt[j];
  int pos = offsets[e] + pre;
  for (int j = 0; j < PER; ++j) {
    int p = base + j;
    if (topi[p] == e) {
      tokIdx[pos]  = p >> 2;               // token
      tokW[pos]    = topv[p];
      pairPos[p]   = pos;
      ++pos;
    }
  }
}

// ---------------- Gate/Up GEMM + SwiGLU -> act (bf16) ----------------
__global__ __launch_bounds__(256) void gateup_kernel(
    const float* __restrict__ x,
    const float* __restrict__ w_gate, const float* __restrict__ w_up,
    const int* __restrict__ counts, const int* __restrict__ offsets,
    const int* __restrict__ tokIdx, unsigned short* __restrict__ act) {
  const int ex = blockIdx.z;
  const int Ne = counts[ex];
  const int mBase = blockIdx.x * BM;       // m fastest: weight panel shared in L2
  if (mBase >= Ne) return;                 // uniform exit, EXEC stays all-ones
  const int n0  = blockIdx.y * BN;
  const int off = offsets[ex];

  __shared__ __align__(16) unsigned short sA[BM][BK + 8];   // row = 80B
  __shared__ __align__(16) unsigned short sG[BK][BN + 8];   // row = 144B
  __shared__ __align__(16) unsigned short sU[BK][BN + 8];

  const int tid = threadIdx.x;
  // --- staging coords (branch-free; clamp rows; garbage rows never stored) ---
  const int arow = tid >> 3, acol = (tid & 7) << 2;         // A: 2 x float4 / thread
  const int tokA0 = tokIdx[off + min(mBase + arow,      Ne - 1)];
  const int tokA1 = tokIdx[off + min(mBase + arow + 32, Ne - 1)];
  const float* xp0 = x + (size_t)tokA0 * H_ + acol;
  const float* xp1 = x + (size_t)tokA1 * H_ + acol;
  const int brow = tid >> 4, bcol = (tid & 15) << 2;        // B: 2 x float4 / matrix
  const float* wg = w_gate + (size_t)ex * H_ * I_ + (size_t)brow * I_ + n0 + bcol;
  const float* wu = w_up   + (size_t)ex * H_ * I_ + (size_t)brow * I_ + n0 + bcol;

  const int lane = tid & 31, wv2 = tid >> 5;
  const int wm = wv2 & 3, wn = wv2 >> 2;
  const int mRow = wm * 16 + (lane & 15);
  const int kb   = (lane < 16) ? 0 : 8;

  v8f accG[2] = {{0,0,0,0,0,0,0,0},{0,0,0,0,0,0,0,0}};
  v8f accU[2] = {{0,0,0,0,0,0,0,0},{0,0,0,0,0,0,0,0}};

  for (int k0 = 0; k0 < H_; k0 += BK) {
    __syncthreads();
    float4 a0 = *(const float4*)(xp0 + k0);
    float4 a1 = *(const float4*)(xp1 + k0);
    float4 g0 = *(const float4*)(wg + (size_t)k0 * I_);
    float4 g1 = *(const float4*)(wg + (size_t)(k0 + 16) * I_);
    float4 u0 = *(const float4*)(wu + (size_t)k0 * I_);
    float4 u1 = *(const float4*)(wu + (size_t)(k0 + 16) * I_);
    uint2 t2;
    t2.x = pk2(a0.x, a0.y); t2.y = pk2(a0.z, a0.w); *(uint2*)&sA[arow][acol]        = t2;
    t2.x = pk2(a1.x, a1.y); t2.y = pk2(a1.z, a1.w); *(uint2*)&sA[arow + 32][acol]   = t2;
    t2.x = pk2(g0.x, g0.y); t2.y = pk2(g0.z, g0.w); *(uint2*)&sG[brow][bcol]        = t2;
    t2.x = pk2(g1.x, g1.y); t2.y = pk2(g1.z, g1.w); *(uint2*)&sG[brow + 16][bcol]   = t2;
    t2.x = pk2(u0.x, u0.y); t2.y = pk2(u0.z, u0.w); *(uint2*)&sU[brow][bcol]        = t2;
    t2.x = pk2(u1.x, u1.y); t2.y = pk2(u1.z, u1.w); *(uint2*)&sU[brow + 16][bcol]   = t2;
    __syncthreads();

    FragAB a, bg0, bg1, bu0, bu1;
    a.q[0] = *(const uint4*)&sA[mRow][kb];
    a.q[1] = *(const uint4*)&sA[mRow][16 + kb];
    const int cb = wn * 32;
    const int nB = cb + (lane & 15);
#pragma unroll
    for (int s = 0; s < 16; ++s) {
      const int kk = (s < 8) ? (kb + s) : (16 + kb + s - 8);
      bg0.v[s] = __builtin_bit_cast(__bf16, sG[kk][nB]);
      bg1.v[s] = __builtin_bit_cast(__bf16, sG[kk][nB + 16]);
      bu0.v[s] = __builtin_bit_cast(__bf16, sU[kk][nB]);
      bu1.v[s] = __builtin_bit_cast(__bf16, sU[kk][nB + 16]);
    }

    accG[0] = __builtin_amdgcn_wmma_f32_16x16x32_bf16(false, a.v, false, bg0.v, (short)0, accG[0], false, false);
    accG[1] = __builtin_amdgcn_wmma_f32_16x16x32_bf16(false, a.v, false, bg1.v, (short)0, accG[1], false, false);
    accU[0] = __builtin_amdgcn_wmma_f32_16x16x32_bf16(false, a.v, false, bu0.v, (short)0, accU[0], false, false);
    accU[1] = __builtin_amdgcn_wmma_f32_16x16x32_bf16(false, a.v, false, bu1.v, (short)0, accU[1], false, false);
  }

#pragma unroll
  for (int s2 = 0; s2 < 2; ++s2) {
#pragma unroll
    for (int i = 0; i < 8; ++i) {
      int m = (lane < 16) ? i : (i + 8);
      int r = mBase + wm * 16 + m;
      if (r < Ne) {
        int col = n0 + wn * 32 + s2 * 16 + (lane & 15);
        float g = accG[s2][i], u = accU[s2][i];
        // silu via fast v_rcp_f32 (1 ulp) instead of IEEE division
        float av = g * __builtin_amdgcn_rcpf(1.f + __expf(-g)) * u;
        act[(size_t)(off + r) * I_ + col] = f2bf(av);
      }
    }
  }
}

// ---------------- Down GEMM; per-pair output (or atomic fallback) ------------
__global__ __launch_bounds__(256) void down_kernel(
    const unsigned short* __restrict__ act, const float* __restrict__ w_down,
    const int* __restrict__ counts, const int* __restrict__ offsets,
    const int* __restrict__ tokIdx, const float* __restrict__ tokW,
    float* __restrict__ pairOut, float* __restrict__ out, int usePairOut) {
  const int ex = blockIdx.z;
  const int Ne = counts[ex];
  const int mBase = blockIdx.x * BM;
  if (mBase >= Ne) return;
  const int n0  = blockIdx.y * BN;
  const int off = offsets[ex];

  __shared__ int   rowTok[BM];
  __shared__ float rowW[BM];
  __shared__ __align__(16) unsigned short sA[BM][BK + 8];
  __shared__ __align__(16) unsigned short sB[BK][BN + 8];

  const int tid = threadIdx.x;
  if (tid < BM) {
    int r = min(mBase + tid, Ne - 1);
    rowTok[tid] = tokIdx[off + r];
    rowW[tid]   = (mBase + tid < Ne) ? tokW[off + r] : 0.f;
  }

  // A tile: act rows are bf16 in memory -> one b128 per thread per K-step
  const int drow = tid >> 2, dpart = (tid & 3) << 3;        // dpart: ushort offset
  const size_t abase = (size_t)(off + min(mBase + drow, Ne - 1)) * I_ + dpart;
  // B tile: fp32 w_down -> bf16
  const int brow = tid >> 4, bcol = (tid & 15) << 2;
  const float* wd = w_down + (size_t)ex * I_ * H_ + (size_t)brow * H_ + n0 + bcol;

  const int lane = tid & 31, wv2 = tid >> 5;
  const int wm = wv2 & 3, wn = wv2 >> 2;
  const int mRow = wm * 16 + (lane & 15);
  const int kb   = (lane < 16) ? 0 : 8;

  v8f acc[2] = {{0,0,0,0,0,0,0,0},{0,0,0,0,0,0,0,0}};

  for (int k0 = 0; k0 < I_; k0 += BK) {
    __syncthreads();
#ifdef HAS_ASYNC_LDS
    __builtin_amdgcn_global_load_async_to_lds_b128(
        (AS1 i4*)(unsigned long long)(act + abase + k0),
        (AS3 i4*)(void*)&sA[drow][dpart], 0, 0);
#else
    *(uint4*)&sA[drow][dpart] = *(const uint4*)(act + abase + k0);
#endif
    float4 b0v = *(const float4*)(wd + (size_t)k0 * H_);
    float4 b1v = *(const float4*)(wd + (size_t)(k0 + 16) * H_);
    uint2 t2;
    t2.x = pk2(b0v.x, b0v.y); t2.y = pk2(b0v.z, b0v.w); *(uint2*)&sB[brow][bcol]      = t2;
    t2.x = pk2(b1v.x, b1v.y); t2.y = pk2(b1v.z, b1v.w); *(uint2*)&sB[brow + 16][bcol] = t2;
    wait_async0();
    __syncthreads();

    FragAB a, b0, b1;
    a.q[0] = *(const uint4*)&sA[mRow][kb];
    a.q[1] = *(const uint4*)&sA[mRow][16 + kb];
    const int cb = wn * 32;
    const int nB = cb + (lane & 15);
#pragma unroll
    for (int s = 0; s < 16; ++s) {
      const int kk = (s < 8) ? (kb + s) : (16 + kb + s - 8);
      b0.v[s] = __builtin_bit_cast(__bf16, sB[kk][nB]);
      b1.v[s] = __builtin_bit_cast(__bf16, sB[kk][nB + 16]);
    }
    acc[0] = __builtin_amdgcn_wmma_f32_16x16x32_bf16(false, a.v, false, b0.v, (short)0, acc[0], false, false);
    acc[1] = __builtin_amdgcn_wmma_f32_16x16x32_bf16(false, a.v, false, b1.v, (short)0, acc[1], false, false);
  }

#pragma unroll
  for (int s2 = 0; s2 < 2; ++s2) {
#pragma unroll
    for (int i = 0; i < 8; ++i) {
      int m = (lane < 16) ? i : (i + 8);
      int rl = wm * 16 + m;
      int r  = mBase + rl;
      if (r < Ne) {
        int col = n0 + wn * 32 + s2 * 16 + (lane & 15);
        float v = acc[s2][i] * rowW[rl];
        if (usePairOut) pairOut[(size_t)(off + r) * H_ + col] = v;
        else            atomicAdd(&out[(size_t)rowTok[rl] * H_ + col], v);
      }
    }
  }
}

// ------------- Fixed-order combine: out[t,h] = sum_k pairOut[pos(t,k),h] -----
__global__ void combine_kernel(const float* __restrict__ pairOut,
                               const int* __restrict__ pairPos,
                               float* __restrict__ out) {
  int idx = blockIdx.x * 256 + threadIdx.x;
  if (idx >= T_ * H_) return;
  int t = idx >> 11;            // / H_
  int h = idx & (H_ - 1);
  const int* pp = pairPos + t * TOPK;
  float s = 0.f;
#pragma unroll
  for (int k = 0; k < TOPK; ++k) s += pairOut[(size_t)pp[k] * H_ + h];
  out[idx] = s;
}

extern "C" void kernel_launch(void* const* d_in, const int* in_sizes, int n_in,
                              void* d_out, int out_size, void* d_ws, size_t ws_size,
                              hipStream_t stream) {
  const float* x      = (const float*)d_in[0];
  const float* gate_w = (const float*)d_in[1];
  const float* w_gate = (const float*)d_in[2];
  const float* w_up   = (const float*)d_in[3];
  const float* w_down = (const float*)d_in[4];

  char* ws = (char*)d_ws;
  int*   topi    = (int*)(ws + OFF_TOPI);
  float* topv    = (float*)(ws + OFF_TOPV);
  int*   counts  = (int*)(ws + OFF_COUNTS);
  int*   offs    = (int*)(ws + OFF_OFFS);
  int*   pairPos = (int*)(ws + OFF_PAIRPOS);
  int*   tokIdx  = (int*)(ws + OFF_TOKIDX);
  float* tokW    = (float*)(ws + OFF_TOKW);
  unsigned short* act = (unsigned short*)(ws + OFF_ACT);
  float* pairOut = (float*)(ws + OFF_PAIROUT);
  float* out = (float*)d_out;

  const int usePairOut = (ws_size >= WS_NEED) ? 1 : 0;

  (void)hipMemsetAsync(counts, 0, E_ * sizeof(int), stream);
  if (!usePairOut)
    (void)hipMemsetAsync(out, 0, (size_t)out_size * sizeof(float), stream);

  router_kernel<<<T_ / 8, 256, 0, stream>>>(x, gate_w, topi, topv, counts);
  scan_kernel<<<1, 32, 0, stream>>>(counts, offs);
  scatter_kernel<<<E_, 256, 0, stream>>>(topi, topv, offs, tokIdx, tokW, pairPos);

  dim3 gA(T_ / BM, I_ / BN, E_);     // m fastest -> weight panels shared in L2
  gateup_kernel<<<gA, 256, 0, stream>>>(x, w_gate, w_up, counts, offs, tokIdx, act);
  dim3 gB(T_ / BM, H_ / BN, E_);
  down_kernel<<<gB, 256, 0, stream>>>(act, w_down, counts, offs, tokIdx, tokW,
                                      pairOut, out, usePairOut);
  if (usePairOut)
    combine_kernel<<<(T_ * H_ + 255) / 256, 256, 0, stream>>>(pairOut, pairPos, out);
}
